// InverseDistanceAttention_38414187495506
// MI455X (gfx1250) — matrix operand order repaired and checked by reading, not compiled, not run on hardware
//
#include <hip/hip_runtime.h>
#include <math.h>

// ---------------- problem constants ----------------
constexpr int kB = 32, kS = 48, kL = 64, kF = 256;
constexpr int kSChunk = 8;            // s-values processed per workgroup
constexpr int XROW = kF + 4;          // f32 LDS row pitch (+16B pad -> conflict-free frag loads)
constexpr int WROW = kF + 8;          // bf16 LDS row pitch (+16B pad)

typedef __bf16 bf16;
typedef __attribute__((ext_vector_type(16))) __bf16 v16bf;
typedef __attribute__((ext_vector_type(8)))  __bf16 v8bf;
typedef __attribute__((ext_vector_type(8)))  float  v8f;
typedef __attribute__((ext_vector_type(4)))  float  v4f;
typedef int v4i __attribute__((vector_size(16)));   // matches builtin param type

#if defined(__has_builtin)
#if __has_builtin(__builtin_amdgcn_global_load_async_to_lds_b128) && \
    __has_builtin(__builtin_amdgcn_s_wait_asynccnt)
#define HAS_ASYNC_LDS 1
#endif
#endif
#ifndef HAS_ASYNC_LDS
#define HAS_ASYNC_LDS 0
#endif

__global__ __launch_bounds__(256) void ida_fused_kernel(
    const float* __restrict__ features,   // (B,S,L,F)
    const float* __restrict__ src_locs,   // (B,L,2)
    const float* __restrict__ tar_locs,   // (B,2)
    const float* __restrict__ W1,         // (F,F) row-major [k][n]
    const float* __restrict__ b1,         // (F)
    const float* __restrict__ W2,         // (F,1)
    const float* __restrict__ b2,         // (1)
    float* __restrict__ out)              // (B,S,F)
{
  __shared__ float Xf[2][kL * XROW];     // double-buffered feature tile, f32
  __shared__ bf16  W1T[kF * WROW];       // W1 transposed [n][k], bf16
  __shared__ float invd_s[kL];
  __shared__ float score_s[kL];
  __shared__ float tl_s[kL];
  __shared__ float el_s[kL];

  const int tid  = threadIdx.x;
  const int lane = tid & 31;
  const int w    = tid >> 5;             // wave id 0..7
  const int b    = blockIdx.y;
  const int s0   = blockIdx.x * kSChunk;

  // ---- stage W1^T as bf16 into LDS (coalesced f32 reads, scattered b16 stores) ----
  for (int idx = tid * 4; idx < kF * kF; idx += 256 * 4) {
    const int k = idx >> 8, n = idx & 255;
    v4f wv = *(const v4f*)(W1 + idx);
    W1T[(n + 0) * WROW + k] = (bf16)wv.x;
    W1T[(n + 1) * WROW + k] = (bf16)wv.y;
    W1T[(n + 2) * WROW + k] = (bf16)wv.z;
    W1T[(n + 3) * WROW + k] = (bf16)wv.w;
  }
  // ---- inverse distances (per b, per l) ----
  if (tid < kL) {
    const float dx = src_locs[(b * kL + tid) * 2 + 0] - tar_locs[b * 2 + 0];
    const float dy = src_locs[(b * kL + tid) * 2 + 1] - tar_locs[b * 2 + 1];
    invd_s[tid] = 1.0f / sqrtf(dx * dx + dy * dy);
  }

  // ---- per-lane epilogue constants (bias + W2 columns this lane owns) ----
  const int mt  = w & 3;                 // M tile (rows mt*16..mt*16+15)
  const int ntg = w >> 2;                // N tile group (8 tiles each)
  float b1v[8], w2v[8];
#pragma unroll
  for (int j = 0; j < 8; ++j) {
    const int n = (ntg * 8 + j) * 16 + (lane & 15);
    b1v[j] = b1[n];
    w2v[j] = W2[n];
  }
  const float b2v = b2[0];

  const float* fbase = features + (size_t)(b * kS + s0) * kL * kF;

  // ---- feature-tile loader: 16 x B128 per thread ----
  auto issue_tile = [&](int buf, int i) {
    const float* src = fbase + (size_t)i * (kL * kF);
#pragma unroll
    for (int r = 0; r < 16; ++r) {
      const int q = tid + 256 * r;       // float4 index within tile
      const int e = q * 4;
      const int l = e >> 8, f = e & 255;
      float* dst = &Xf[buf][l * XROW + f];
#if HAS_ASYNC_LDS
      __builtin_amdgcn_global_load_async_to_lds_b128(
          (__attribute__((address_space(1))) v4i*)(src + e),
          (__attribute__((address_space(3))) v4i*)dst, 0, 0);
#else
      *(v4f*)dst = *(const v4f*)(src + e);
#endif
    }
  };
  auto wait_tiles = [&]() {
#if HAS_ASYNC_LDS
    __builtin_amdgcn_s_wait_asynccnt(0);
#endif
  };

  issue_tile(0, 0);
  __syncthreads();                       // W1T + invd staged

  const int mrow   = mt * 16 + (lane & 15);   // A: lane = row within M tile
  const int khalfA = (lane >> 4) * 8;         // A: K sub-chunk base per lane half
  const int khalfB = (lane >> 4) * 8;         // B: K sub-chunk base per lane half
  const int nloc   = lane & 15;

  for (int i = 0; i < kSChunk; ++i) {
    wait_tiles();
    __syncthreads();                     // tile i resident; prev iter fully consumed
    if (tid < kL) score_s[tid] = 0.0f;
    if (i + 1 < kSChunk) issue_tile((i + 1) & 1, i + 1);
    __syncthreads();                     // score zeroed

    const float* xb = &Xf[i & 1][0];

    // ---------------- GEMM1: H = relu(X * W1 + b1), via bf16 WMMA ----------------
    v8f acc[8];
    const v8f zero = {0.f, 0.f, 0.f, 0.f, 0.f, 0.f, 0.f, 0.f};
#pragma unroll
    for (int j = 0; j < 8; ++j) acc[j] = zero;

#pragma unroll
    for (int kt = 0; kt < 8; ++kt) {
      const int K0 = kt * 32;
      // A fragment: rows of X, two contiguous 8-float K chunks per lane
      const float* ap = xb + mrow * XROW + K0 + khalfA;
      v4f a0 = *(const v4f*)(ap + 0);
      v4f a1 = *(const v4f*)(ap + 4);
      v4f a2 = *(const v4f*)(ap + 16);
      v4f a3 = *(const v4f*)(ap + 20);
      float at[16] = {a0.x, a0.y, a0.z, a0.w, a1.x, a1.y, a1.z, a1.w,
                      a2.x, a2.y, a2.z, a2.w, a3.x, a3.y, a3.z, a3.w};
      v16bf af;
#pragma unroll
      for (int t = 0; t < 16; ++t) af[t] = (bf16)at[t];

#pragma unroll
      for (int j = 0; j < 8; ++j) {
        const int n = (ntg * 8 + j) * 16 + nloc;
        const bf16* bp = &W1T[n * WROW + K0 + khalfB];
        v8bf blo = *(const v8bf*)(bp);
        v8bf bhi = *(const v8bf*)(bp + 16);
        v16bf bb = __builtin_shufflevector(blo, bhi, 0, 1, 2, 3, 4, 5, 6, 7,
                                           8, 9, 10, 11, 12, 13, 14, 15);
        acc[j] = __builtin_amdgcn_wmma_f32_16x16x32_bf16(
            false, af, false, bb, (short)0, acc[j], false, false);
      }
    }

    // ---- fused epilogue: relu(+b1), dot with W2, reduce over N to per-row score ----
#pragma unroll
    for (int v = 0; v < 8; ++v) {
      float sum = 0.0f;
#pragma unroll
      for (int j = 0; j < 8; ++j) {
        float c = acc[j][v] + b1v[j];
        c = c > 0.0f ? c : 0.0f;
        sum += c * w2v[j];
      }
      sum += __shfl_xor(sum, 1, 32);
      sum += __shfl_xor(sum, 2, 32);
      sum += __shfl_xor(sum, 4, 32);
      sum += __shfl_xor(sum, 8, 32);
      const int row = mt * 16 + v + (lane >> 4) * 8;  // C layout: lanes 16-31 -> M+8
      if ((lane & 15) == 0) atomicAdd(&score_s[row], sum);
    }
    __syncthreads();                     // scores complete

    // ---- t[l] = relu(score + b2) * invd[l] ----
    if (tid < kL) {
      float t = score_s[tid] + b2v;
      t = t > 0.0f ? t : 0.0f;
      tl_s[tid] = t * invd_s[tid];
    }
    __syncthreads();

    // ---- softmax over L (broadcast LDS reads, redundant per-thread scalars) ----
    float mx = -3.402823466e38f;
#pragma unroll 8
    for (int l = 0; l < kL; ++l) mx = fmaxf(mx, tl_s[l]);
    if (tid < kL) el_s[tid] = __expf(tl_s[tid] - mx);
    __syncthreads();

    float ssum = 0.0f;
#pragma unroll 8
    for (int l = 0; l < kL; ++l) ssum += el_s[l];

    // ---- out[b,s,f] = sum_l X[l][f] * e[l] / ssum   (f = tid) ----
    float accum = 0.0f;
#pragma unroll 8
    for (int l = 0; l < kL; ++l) accum += xb[l * XROW + tid] * el_s[l];
    out[(size_t)(b * kS + (s0 + i)) * kF + tid] = accum / ssum;
  }
}

extern "C" void kernel_launch(void* const* d_in, const int* in_sizes, int n_in,
                              void* d_out, int out_size, void* d_ws, size_t ws_size,
                              hipStream_t stream) {
  const float* features = (const float*)d_in[0];
  const float* src_locs = (const float*)d_in[1];
  const float* tar_locs = (const float*)d_in[2];
  const float* W1       = (const float*)d_in[3];
  const float* b1       = (const float*)d_in[4];
  const float* W2       = (const float*)d_in[5];
  const float* b2       = (const float*)d_in[6];
  float* out = (float*)d_out;

  dim3 grid(kS / kSChunk, kB);           // 6 x 32 = 192 workgroups
  ida_fused_kernel<<<grid, 256, 0, stream>>>(features, src_locs, tar_locs,
                                             W1, b1, W2, b2, out);
}
